// transformer_65377992179818
// MI455X (gfx1250) — matrix operand order repaired and checked by reading, not compiled, not run on hardware
//
#include <hip/hip_runtime.h>

// ---------------------------------------------------------------------------
// Model constants (from reference)
// ---------------------------------------------------------------------------
#define BZ     4
#define NN     2048
#define DIMC   768
#define HEADSC 12
#define DHC    64
#define DEPTHC 4
#define WC     35
#define PC     384     // DIM/2
#define V4C    192     // DIM/4
#define PBC    192     // DIM/4
#define FFC    3072    // DIM*4
#define NWC    59      // ceil(N/W)
#define NPC    (NWC*WC)  // 2065 padded seq length
#define NEGF   3.402823466e38f

typedef __attribute__((ext_vector_type(16))) __bf16        bf16x16;
typedef __attribute__((ext_vector_type(8)))  float         floatx8;
typedef __attribute__((ext_vector_type(8)))  unsigned int  uintx8;

// fp32 -> bf16 round-to-nearest-even (bit trick; done once at producers)
__device__ __forceinline__ unsigned short f2bf(float f) {
  unsigned int u = __float_as_uint(f);
  u += 0x7FFFu + ((u >> 16) & 1u);
  return (unsigned short)(u >> 16);
}
__device__ __forceinline__ float bf2f(unsigned short u) {
  return __uint_as_float((unsigned int)u << 16);
}

// ---------------------------------------------------------------------------
// CDNA5 async global->LDS copy (16B per lane), guarded with sync fallback.
// Builtin signature (from hipcc diagnostic): arg0 = AS1 (global) v4i*,
// arg1 = AS3 (LDS) v4i*, then imm offset, imm cpol.
// ---------------------------------------------------------------------------
#if __has_builtin(__builtin_amdgcn_global_load_async_to_lds_b128)
#define HAS_ASYNC 1
typedef int v4i_t __attribute__((vector_size(16)));
typedef __attribute__((address_space(1))) v4i_t as1_v4i;
typedef __attribute__((address_space(3))) v4i_t as3_v4i;
#else
#define HAS_ASYNC 0
#endif

__device__ __forceinline__ void async_copy16(const unsigned short* g, unsigned int* l) {
#if HAS_ASYNC
  __builtin_amdgcn_global_load_async_to_lds_b128((as1_v4i*)g, (as3_v4i*)l, 0, 0);
#else
  *(uint4*)l = *(const uint4*)g;
#endif
}
__device__ __forceinline__ void wait_async3() {
#if HAS_ASYNC
#if __has_builtin(__builtin_amdgcn_s_wait_asynccnt)
  __builtin_amdgcn_s_wait_asynccnt(3);
#else
  asm volatile("s_wait_asynccnt 0x3" ::: "memory");
#endif
#endif
}
__device__ __forceinline__ void wait_async0() {
#if HAS_ASYNC
#if __has_builtin(__builtin_amdgcn_s_wait_asynccnt)
  __builtin_amdgcn_s_wait_asynccnt(0);
#else
  asm volatile("s_wait_asynccnt 0x0" ::: "memory");
#endif
#endif
}

// ---------------------------------------------------------------------------
// bf16 WMMA GEMM:  C[M,N] = epilogue( A[M,K](bf16) @ WtT[N,K](bf16, pre-transposed) )
//   Block tile 128x64 (256 thr = 8 wave32), BK=32, wave tile 32x32
//   -> 4 x v_wmma_f32_16x16x32_bf16 per K step per wave.
//   LDS ping-pong double buffer filled by GLOBAL_LOAD_ASYNC_TO_LDS_B128;
//   OOB / window-pad rows read from a zeroed scratch line (async-compatible).
//   flags: 1=+bias[n], 2=accumulate into C(fp32), 4=relu, 8=store bf16
// ---------------------------------------------------------------------------
__device__ __forceinline__ void stage_tile(
    const unsigned short* __restrict__ A, const unsigned short* __restrict__ Wt,
    const unsigned short* __restrict__ zbuf,
    unsigned int (*As)[20], unsigned int (*Bs)[20],
    int tid, int mBase, int nBase, int M, int K, int N,
    int padRows, int realRows, int k0)
{
  // A: 128 rows x 32 halves = 512 x 16B chunks -> 2 per thread
  for (int s = tid; s < 512; s += 256) {
    int r = s >> 2, q = s & 3;
    int gm = mBase + r;
    const unsigned short* gp = zbuf;
    if (gm < M) {
      long src = gm;
      if (padRows) {                       // window padding: zero rows
        int b  = gm / padRows;
        int nn = gm - b * padRows;
        src = (nn < realRows) ? ((long)b * realRows + nn) : -1;
      }
      if (src >= 0) gp = A + (size_t)src * K + k0 + q * 8;
    }
    async_copy16(gp, &As[r][q * 4]);
  }
  // B: 64 rows (N-major, pre-transposed) x 32 halves = 256 chunks -> 1 per thread
  {
    int r = tid >> 2, q = tid & 3;
    int gn = nBase + r;
    const unsigned short* gp = (gn < N) ? (Wt + (size_t)gn * K + k0 + q * 8) : zbuf;
    async_copy16(gp, &Bs[r][q * 4]);
  }
}

__global__ void __launch_bounds__(256)
gemm_bf16_wmma(const unsigned short* __restrict__ A,
               const unsigned short* __restrict__ Wt,   // [N][K] bf16
               const float* __restrict__ bias,
               void* __restrict__ C,
               const unsigned short* __restrict__ zbuf,
               int M, int K, int N, int padRows, int realRows, int flags)
{
  __shared__ unsigned int As32[2][128][20];  // [m][k/2], stride 20 words
  __shared__ unsigned int Bs32[2][64][20];   // [n][k/2]

  const int tid  = threadIdx.x;
  const int lane = tid & 31;
  const int wv   = tid >> 5;     // 8 waves
  const int wvM  = wv & 3;       // 4 x 32-row bands
  const int wvN  = wv >> 2;      // 2 x 32-col bands
  const int hk   = lane >> 4;
  const int l15  = lane & 15;
  const int mBase = blockIdx.y * 128;
  const int nBase = blockIdx.x * 64;

  floatx8 acc00 = {0.f,0.f,0.f,0.f,0.f,0.f,0.f,0.f};
  floatx8 acc01 = acc00, acc10 = acc00, acc11 = acc00;

  const int KT = K >> 5;
  stage_tile(A, Wt, zbuf, As32[0], Bs32[0], tid, mBase, nBase, M, K, N,
             padRows, realRows, 0);
  int buf = 0;
  for (int kt = 0; kt < KT; ++kt) {
    if (kt + 1 < KT) {
      // overlap: issue next tile into other buffer, then wait only for current
      stage_tile(A, Wt, zbuf, As32[buf ^ 1], Bs32[buf ^ 1], tid, mBase, nBase,
                 M, K, N, padRows, realRows, (kt + 1) << 5);
      wait_async3();
    } else {
      wait_async0();
    }
    __syncthreads();

    uintx8 a0, a1, b0, b1;
    const int am = wvM * 32 + l15;
    const int bn = wvN * 32 + l15;
    #pragma unroll
    for (int v = 0; v < 8; ++v) {
      const int kwA = hk * 8 + v;                       // A: K = half*16 + 2v
      a0[v] = As32[buf][am     ][kwA];
      a1[v] = As32[buf][am + 16][kwA];
      const int kwB = hk * 8 + v;                       // B: K = half*16 + 2v
      b0[v] = Bs32[buf][bn     ][kwB];
      b1[v] = Bs32[buf][bn + 16][kwB];
    }
    acc00 = __builtin_amdgcn_wmma_f32_16x16x32_bf16(false, __builtin_bit_cast(bf16x16, a0),
            false, __builtin_bit_cast(bf16x16, b0), (short)0, acc00, false, false);
    acc01 = __builtin_amdgcn_wmma_f32_16x16x32_bf16(false, __builtin_bit_cast(bf16x16, a0),
            false, __builtin_bit_cast(bf16x16, b1), (short)0, acc01, false, false);
    acc10 = __builtin_amdgcn_wmma_f32_16x16x32_bf16(false, __builtin_bit_cast(bf16x16, a1),
            false, __builtin_bit_cast(bf16x16, b0), (short)0, acc10, false, false);
    acc11 = __builtin_amdgcn_wmma_f32_16x16x32_bf16(false, __builtin_bit_cast(bf16x16, a1),
            false, __builtin_bit_cast(bf16x16, b1), (short)0, acc11, false, false);
    __syncthreads();
    buf ^= 1;
  }

  // epilogue (C/D layout: VGPR v -> m = v + 8*half, n = lane%16)
  #pragma unroll
  for (int mi = 0; mi < 2; ++mi) {
    #pragma unroll
    for (int ni = 0; ni < 2; ++ni) {
      const floatx8 av = (mi == 0) ? (ni == 0 ? acc00 : acc01)
                                   : (ni == 0 ? acc10 : acc11);
      const int gn = nBase + wvN * 32 + ni * 16 + l15;
      if (gn < N) {
        #pragma unroll
        for (int v = 0; v < 8; ++v) {
          const int gm = mBase + wvM * 32 + mi * 16 + hk * 8 + v;
          if (gm < M) {
            float val = av[v];
            if (flags & 1) val += bias[gn];
            if (flags & 4) val = fmaxf(val, 0.f);
            const size_t off = (size_t)gm * N + gn;
            if (flags & 8) {
              ((unsigned short*)C)[off] = f2bf(val);
            } else {
              float* Cf = (float*)C;
              if (flags & 2) val += Cf[off];
              Cf[off] = val;
            }
          }
        }
      }
    }
  }
}

// ---------------------------------------------------------------------------
// Weight convert + transpose: in[K][N] fp32 -> out[N][K] bf16 (once per layer)
// ---------------------------------------------------------------------------
__global__ void convtr_kernel(const float* __restrict__ in,
                              unsigned short* __restrict__ out, int K, int N)
{
  long idx = (long)blockIdx.x * 256 + threadIdx.x;
  if (idx >= (long)K * N) return;
  int n = (int)(idx / K);
  int k = (int)(idx - (long)n * K);
  out[idx] = f2bf(in[(size_t)k * N + n]);
}

// ---------------------------------------------------------------------------
// LayerNorm over DIM=768 -> bf16 output, one 256-thread block per row
// ---------------------------------------------------------------------------
__global__ void __launch_bounds__(256)
ln_kernel(const float* __restrict__ x, const float* __restrict__ g,
          const float* __restrict__ b, unsigned short* __restrict__ out)
{
  __shared__ float s1[256], s2[256];
  const int t = threadIdx.x;
  const float* xr = x + (size_t)blockIdx.x * DIMC;
  float v0 = xr[t], v1 = xr[t + 256], v2 = xr[t + 512];
  s1[t] = v0 + v1 + v2;
  s2[t] = v0 * v0 + v1 * v1 + v2 * v2;
  __syncthreads();
  for (int st = 128; st > 0; st >>= 1) {
    if (t < st) { s1[t] += s1[t + st]; s2[t] += s2[t + st]; }
    __syncthreads();
  }
  const float mean = s1[0] * (1.f / DIMC);
  const float var  = s2[0] * (1.f / DIMC) - mean * mean;
  const float inv  = rsqrtf(var + 1e-5f);
  unsigned short* orow = out + (size_t)blockIdx.x * DIMC;
  orow[t      ] = f2bf((v0 - mean) * inv * g[t      ] + b[t      ]);
  orow[t + 256] = f2bf((v1 - mean) * inv * g[t + 256] + b[t + 256]);
  orow[t + 512] = f2bf((v2 - mean) * inv * g[t + 512] + b[t + 512]);
}

// ---------------------------------------------------------------------------
// Windowed attention (35x35 per head, <2% of FLOPs -> VALU). o stored bf16.
// ---------------------------------------------------------------------------
__global__ void __launch_bounds__(64)
attn_kernel(const float* __restrict__ qkv, const float* __restrict__ bTab,
            const unsigned char* __restrict__ mask, const float* __restrict__ temp,
            int layer, unsigned short* __restrict__ o)
{
  __shared__ float qs[WC][DHC], ks[WC][DHC], vs[WC][DHC];
  __shared__ unsigned char validv[WC];
  const int blk = blockIdx.x;
  const int h  = blk % HEADSC;
  const int bw = blk / HEADSC;
  const int w  = bw % NWC;
  const int b  = bw / NWC;
  const int tid = threadIdx.x;

  for (int idx = tid; idx < WC * DHC; idx += 64) {
    int i = idx >> 6, d = idx & 63;
    size_t base = ((size_t)(b * NPC + w * WC + i)) * (3 * HEADSC * DHC)
                + (size_t)(h * DHC + d) * 3;
    qs[i][d] = qkv[base];
    ks[i][d] = qkv[base + 1];
    vs[i][d] = qkv[base + 2];
  }
  if (tid < WC) {
    int n = w * WC + tid;
    validv[tid] = (n < NN) && (mask[(size_t)b * NN + n] == 0);
  }
  __syncthreads();
  if (tid < WC) {  // l2-normalize q,k rows
    float nq = 0.f, nk = 0.f;
    for (int d = 0; d < DHC; ++d) { nq += qs[tid][d] * qs[tid][d]; nk += ks[tid][d] * ks[tid][d]; }
    float iq = 1.f / fmaxf(sqrtf(nq), 1e-12f);
    float ik = 1.f / fmaxf(sqrtf(nk), 1e-12f);
    for (int d = 0; d < DHC; ++d) { qs[tid][d] *= iq; ks[tid][d] *= ik; }
  }
  __syncthreads();
  if (tid < WC) {
    const int i = tid;
    const float tmp = temp[layer];
    float sc[WC];
    float mx = -NEGF;
    for (int j = 0; j < WC; ++j) {
      bool masked = (j > i) || !(validv[i] && validv[j]);
      float s;
      if (masked) s = -NEGF;
      else {
        float d0 = 0.f;
        for (int d = 0; d < DHC; ++d) d0 += qs[i][d] * ks[j][d];
        s = d0 * tmp + bTab[(h * WC + i) * WC + j];
      }
      sc[j] = s;
      mx = fmaxf(mx, s);
    }
    float sum = 0.f;
    for (int j = 0; j < WC; ++j) { float e = __expf(sc[j] - mx); sc[j] = e; sum += e; }
    const float inv = 1.f / sum;
    size_t orow = ((size_t)(b * NPC + w * WC + i)) * DIMC + (size_t)h * DHC;
    for (int d = 0; d < DHC; ++d) {
      float a = 0.f;
      for (int j = 0; j < WC; ++j) a += sc[j] * vs[j][d];
      o[orow + d] = f2bf(a * inv);
    }
  }
}

// window mean of o over valid positions (1/len folded here; relu GEMM after)
__global__ void __launch_bounds__(256)
omean_kernel(const unsigned short* __restrict__ o,
             const unsigned char* __restrict__ mask, unsigned short* __restrict__ om)
{
  __shared__ unsigned char validv[WC];
  __shared__ float lenf;
  const int bw = blockIdx.x;
  const int b = bw / NWC, w = bw % NWC;
  const int tid = threadIdx.x;
  if (tid < WC) {
    int n = w * WC + tid;
    validv[tid] = (n < NN) && (mask[(size_t)b * NN + n] == 0);
  }
  __syncthreads();
  if (tid == 0) {
    int c = 0;
    for (int j = 0; j < WC; ++j) c += validv[j];
    lenf = (float)(c > 0 ? c : 1);
  }
  __syncthreads();
  const float invl = 1.f / lenf;
  for (int d = tid; d < DIMC; d += 256) {
    float s = 0.f;
    for (int j = 0; j < WC; ++j)
      if (validv[j]) s += bf2f(o[((size_t)(b * NPC + w * WC + j)) * DIMC + d]);
    om[(size_t)bw * DIMC + d] = f2bf(s * invl);
  }
}

// k2 += sinusoid(i) * we_scale[layer]
__global__ void sinu_add_kernel(float* __restrict__ k2,
                                const float* __restrict__ we_scale, int layer)
{
  const long total = (long)BZ * NPC * PC;
  long idx = (long)blockIdx.x * 256 + threadIdx.x;
  if (idx >= total) return;
  int row = (int)(idx / PC);
  int p   = (int)(idx - (long)row * PC);
  int i   = (row % NPC) % WC;
  int q   = (p < V4C) ? p : (p - V4C);
  float ang = (float)i * __powf(10000.f, -(float)q / (float)V4C);
  float v = (p < V4C) ? __sinf(ang) : __cosf(ang);
  k2[idx] += v * we_scale[layer];
}

// 1xW window-summary attention: d2 = wq.k2/sqrt(P), mask, softmax, o2 = p@v2
__global__ void __launch_bounds__(192)
d2o2_kernel(const float* __restrict__ wq, const float* __restrict__ k2,
            const float* __restrict__ v2, const unsigned char* __restrict__ mask,
            float* __restrict__ o2)
{
  __shared__ float p[WC];
  const int bw = blockIdx.x;
  const int b = bw / NWC, w = bw % NWC;
  const int tid = threadIdx.x;
  if (tid < WC) {
    int n = w * WC + tid;
    bool valid = (n < NN) && (mask[(size_t)b * NN + n] == 0);
    float s = -NEGF;
    if (valid) {
      float d0 = 0.f;
      const float* wqp = wq + (size_t)bw * PC;
      const float* kp  = k2 + ((size_t)(b * NPC + w * WC + tid)) * PC;
      for (int q = 0; q < PC; ++q) d0 += wqp[q] * kp[q];
      s = d0 * 0.05103103630798287f;  // 384^-0.5
    }
    p[tid] = s;
  }
  __syncthreads();
  if (tid == 0) {
    float mx = -NEGF;
    for (int j = 0; j < WC; ++j) mx = fmaxf(mx, p[j]);
    float sum = 0.f;
    for (int j = 0; j < WC; ++j) { p[j] = __expf(p[j] - mx); sum += p[j]; }
    float inv = 1.f / sum;
    for (int j = 0; j < WC; ++j) p[j] *= inv;
  }
  __syncthreads();
  float a = 0.f;
  for (int j = 0; j < WC; ++j)
    a += p[j] * v2[((size_t)(b * NPC + w * WC + j)) * V4C + tid];
  o2[(size_t)bw * V4C + tid] = a;
}

// ocat(bf16) = [ o(768, already bf16) | w==0 ? bos : o2[w-1] (192) ]
__global__ void concat_kernel(const unsigned short* __restrict__ o,
                              const float* __restrict__ o2,
                              const float* __restrict__ bos,
                              unsigned short* __restrict__ ocat)
{
  const long total = (long)BZ * NPC * (DIMC + V4C);
  long idx = (long)blockIdx.x * 256 + threadIdx.x;
  if (idx >= total) return;
  long row = idx / (DIMC + V4C);
  int  c   = (int)(idx - row * (DIMC + V4C));
  int  b   = (int)(row / NPC);
  int  rr  = (int)(row - (long)b * NPC);
  int  w   = rr / WC;
  if (c < DIMC) ocat[idx] = o[row * DIMC + c];
  else {
    float v = (w == 0) ? bos[c - DIMC]
                       : o2[((size_t)(b * NWC + w - 1)) * V4C + (c - DIMC)];
    ocat[idx] = f2bf(v);
  }
}

// x[b,n] += xo[b, padded n]   (drop rows beyond N)
__global__ void resadd_kernel(float* __restrict__ x, const float* __restrict__ xo)
{
  const long total = (long)BZ * NN * DIMC;
  long idx = (long)blockIdx.x * 256 + threadIdx.x;
  if (idx >= total) return;
  long row = idx / DIMC;
  int  d   = (int)(idx - row * DIMC);
  int  b   = (int)(row / NN);
  int  n   = (int)(row - (long)b * NN);
  x[idx] += xo[((size_t)b * NPC + n) * DIMC + d];
}

// t(bf16) = a * silu(g), u = [a|g] of width 2*FF
__global__ void glu_kernel(const float* __restrict__ u, unsigned short* __restrict__ t)
{
  const long total = (long)BZ * NN * FFC;
  long idx = (long)blockIdx.x * 256 + threadIdx.x;
  if (idx >= total) return;
  long r = idx / FFC;
  int  f = (int)(idx - r * FFC);
  float a = u[r * (2 * FFC) + f];
  float g = u[r * (2 * FFC) + FFC + f];
  t[idx] = f2bf(a * g / (1.f + __expf(-g)));
}

// ---- relative position bias MLP (once per launch) ----
__global__ void pb1_kernel(const float* __restrict__ W1, const float* __restrict__ b1,
                           float* __restrict__ h1)
{
  int idx = blockIdx.x * 256 + threadIdx.x;
  if (idx >= (2 * WC - 1) * PBC) return;
  int r = idx / PBC, q = idx - r * PBC;
  float pos = (float)(r - (WC - 1));
  h1[idx] = fmaxf(pos * W1[q] + b1[q], 0.f);
}
__global__ void pb2_kernel(const float* __restrict__ h1, const float* __restrict__ W2,
                           const float* __restrict__ b2, float* __restrict__ h2)
{
  int idx = blockIdx.x * 256 + threadIdx.x;
  if (idx >= (2 * WC - 1) * PBC) return;
  int r = idx / PBC, q2 = idx - r * PBC;
  float s = b2[q2];
  for (int q = 0; q < PBC; ++q) s += h1[r * PBC + q] * W2[q * PBC + q2];
  h2[idx] = fmaxf(s, 0.f);
}
__global__ void pb3_kernel(const float* __restrict__ h2, const float* __restrict__ W3,
                           const float* __restrict__ b3, float* __restrict__ bTab)
{
  int idx = blockIdx.x * 256 + threadIdx.x;
  if (idx >= HEADSC * WC * WC) return;
  int h = idx / (WC * WC);
  int rem = idx - h * WC * WC;
  int i = rem / WC, j = rem - i * WC;
  int r = i - j + (WC - 1);
  float s = b3[h];
  for (int q = 0; q < PBC; ++q) s += h2[r * PBC + q] * W3[q * HEADSC + h];
  bTab[idx] = s;
}

// ---------------------------------------------------------------------------
// Host orchestration
// ---------------------------------------------------------------------------
extern "C" void kernel_launch(void* const* d_in, const int* in_sizes, int n_in,
                              void* d_out, int out_size, void* d_ws, size_t ws_size,
                              hipStream_t stream)
{
  (void)in_sizes; (void)n_in; (void)out_size; (void)ws_size;

  const float*         x_in    = (const float*)d_in[0];
  const unsigned char* mask    = (const unsigned char*)d_in[1];
  const float* ln1_g   = (const float*)d_in[2];
  const float* ln1_b   = (const float*)d_in[3];
  const float* qkv_W   = (const float*)d_in[4];
  const float* temp    = (const float*)d_in[5];
  const float* we_lin  = (const float*)d_in[6];
  const float* we_q    = (const float*)d_in[7];
  const float* we_k    = (const float*)d_in[8];
  const float* we_v    = (const float*)d_in[9];
  const float* we_bos  = (const float*)d_in[10];
  const float* we_sc   = (const float*)d_in[11];
  const float* out_W   = (const float*)d_in[12];
  const float* ln2_g   = (const float*)d_in[13];
  const float* ln2_b   = (const float*)d_in[14];
  const float* glu_W   = (const float*)d_in[15];
  const float* glu_b   = (const float*)d_in[16];
  const float* ffo_W   = (const float*)d_in[17];
  const float* ffo_b   = (const float*)d_in[18];
  const float* pb_W1   = (const float*)d_in[19];
  const float* pb_b1   = (const float*)d_in[20];
  const float* pb_W2   = (const float*)d_in[21];
  const float* pb_b2   = (const float*)d_in[22];
  const float* pb_W3   = (const float*)d_in[23];
  const float* pb_b3   = (const float*)d_in[24];

  const int rowsX   = BZ * NN;           // 8192
  const int rowsWin = BZ * NPC;          // 8260
  const int nwB     = BZ * NWC;          // 236
  const int QKVN    = 3 * HEADSC * DHC;  // 2304

  float* x = (float*)d_out;              // residual stream lives in d_out
  (void)hipMemcpyAsync(x, x_in, (size_t)rowsX * DIMC * sizeof(float),
                       hipMemcpyDeviceToDevice, stream);

  // workspace bump allocator
  char* ws = (char*)d_ws;
  size_t off = 0;
  auto alloc = [&](size_t bytes) -> void* {
    off = (off + 255) & ~(size_t)255;
    void* p = (void*)(ws + off);
    off += bytes;
    return p;
  };
  // fp32 intermediates
  float* qkv  = (float*)alloc((size_t)rowsWin * QKVN * 4);
  float* bTab = (float*)alloc((size_t)HEADSC * WC * WC * 4);
  float* h1b  = (float*)alloc((size_t)(2 * WC - 1) * PBC * 4);
  float* h2b  = (float*)alloc((size_t)(2 * WC - 1) * PBC * 4);
  float* wqb  = (float*)alloc((size_t)nwB * PC * 4);
  float* k2b  = (float*)alloc((size_t)rowsWin * PC * 4);
  float* v2b  = (float*)alloc((size_t)rowsWin * V4C * 4);
  float* o2b  = (float*)alloc((size_t)nwB * V4C * 4);
  float* xo   = (float*)alloc((size_t)rowsWin * DIMC * 4);
  float* ub   = (float*)alloc((size_t)rowsX * 2 * FFC * 4);
  // bf16 activations
  unsigned short* h16   = (unsigned short*)alloc((size_t)rowsX * DIMC * 2);
  unsigned short* o16   = (unsigned short*)alloc((size_t)rowsWin * DIMC * 2);
  unsigned short* om16  = (unsigned short*)alloc((size_t)nwB * DIMC * 2);
  unsigned short* web16 = (unsigned short*)alloc((size_t)nwB * PC * 2);
  unsigned short* oc16  = (unsigned short*)alloc((size_t)rowsWin * (DIMC + V4C) * 2);
  unsigned short* tb16  = (unsigned short*)alloc((size_t)rowsX * FFC * 2);
  // bf16 transposed weights (re-converted per layer, reused across M-tiles)
  unsigned short* qkvT  = (unsigned short*)alloc((size_t)DIMC * QKVN * 2);
  unsigned short* welT  = (unsigned short*)alloc((size_t)DIMC * PC * 2);
  unsigned short* weqT  = (unsigned short*)alloc((size_t)PC * PC * 2);
  unsigned short* wekT  = (unsigned short*)alloc((size_t)DIMC * PC * 2);
  unsigned short* wevT  = (unsigned short*)alloc((size_t)DIMC * V4C * 2);
  unsigned short* outT  = (unsigned short*)alloc((size_t)(DIMC + V4C) * DIMC * 2);
  unsigned short* gluT  = (unsigned short*)alloc((size_t)DIMC * 2 * FFC * 2);
  unsigned short* ffoT  = (unsigned short*)alloc((size_t)FFC * DIMC * 2);
  // zero line for async-pad reads
  unsigned short* zb    = (unsigned short*)alloc(256);
  (void)hipMemsetAsync(zb, 0, 256, stream);

  auto convtr = [&](const float* in, unsigned short* out2, int Kd, int Nd) {
    long tot = (long)Kd * Nd;
    convtr_kernel<<<(int)((tot + 255) / 256), 256, 0, stream>>>(in, out2, Kd, Nd);
  };

  // relative-position bias table (layer independent)
  pb1_kernel<<<((2 * WC - 1) * PBC + 255) / 256, 256, 0, stream>>>(pb_W1, pb_b1, h1b);
  pb2_kernel<<<((2 * WC - 1) * PBC + 255) / 256, 256, 0, stream>>>(h1b, pb_W2, pb_b2, h2b);
  pb3_kernel<<<(HEADSC * WC * WC + 255) / 256, 256, 0, stream>>>(h2b, pb_W3, pb_b3, bTab);

  for (int l = 0; l < DEPTHC; ++l) {
    // weights -> bf16, transposed [N][K] (one pass; GEMMs then stream raw 16B)
    convtr(qkv_W + (size_t)l * DIMC * QKVN,        qkvT, DIMC, QKVN);
    convtr(we_lin + (size_t)l * DIMC * PC,         welT, DIMC, PC);
    convtr(we_q  + (size_t)l * PC * PC,            weqT, PC,   PC);
    convtr(we_k  + (size_t)l * DIMC * PC,          wekT, DIMC, PC);
    convtr(we_v  + (size_t)l * DIMC * V4C,         wevT, DIMC, V4C);
    convtr(out_W + (size_t)l * (DIMC + V4C) * DIMC, outT, DIMC + V4C, DIMC);
    convtr(glu_W + (size_t)l * DIMC * 2 * FFC,     gluT, DIMC, 2 * FFC);
    convtr(ffo_W + (size_t)l * FFC * DIMC,         ffoT, FFC,  DIMC);

    // ---- attention branch ----
    ln_kernel<<<rowsX, 256, 0, stream>>>(x, ln1_g + l * DIMC, ln1_b + l * DIMC, h16);
    gemm_bf16_wmma<<<dim3(QKVN / 64, (rowsWin + 127) / 128), 256, 0, stream>>>(
        h16, qkvT, nullptr, qkv, zb, rowsWin, DIMC, QKVN, NPC, NN, 0);
    attn_kernel<<<BZ * NWC * HEADSC, 64, 0, stream>>>(qkv, bTab, mask, temp, l, o16);

    // ---- window-summary branch ----
    omean_kernel<<<nwB, 256, 0, stream>>>(o16, mask, om16);
    gemm_bf16_wmma<<<dim3(PC / 64, (nwB + 127) / 128), 256, 0, stream>>>(
        om16, welT, nullptr, web16, zb, nwB, DIMC, PC, 0, 0, 4 | 8);
    gemm_bf16_wmma<<<dim3(PC / 64, (nwB + 127) / 128), 256, 0, stream>>>(
        web16, weqT, nullptr, wqb, zb, nwB, PC, PC, 0, 0, 0);
    gemm_bf16_wmma<<<dim3(PC / 64, (rowsWin + 127) / 128), 256, 0, stream>>>(
        o16, wekT, nullptr, k2b, zb, rowsWin, DIMC, PC, 0, 0, 0);
    sinu_add_kernel<<<(int)(((long)rowsWin * PC + 255) / 256), 256, 0, stream>>>(
        k2b, we_sc, l);
    gemm_bf16_wmma<<<dim3(V4C / 64, (rowsWin + 127) / 128), 256, 0, stream>>>(
        o16, wevT, nullptr, v2b, zb, rowsWin, DIMC, V4C, 0, 0, 0);
    d2o2_kernel<<<nwB, 192, 0, stream>>>(wqb, k2b, v2b, mask, o2b);

    // ---- merge + output projection + residual ----
    concat_kernel<<<(int)(((long)rowsWin * (DIMC + V4C) + 255) / 256), 256, 0, stream>>>(
        o16, o2b, we_bos + (size_t)l * V4C, oc16);
    gemm_bf16_wmma<<<dim3(DIMC / 64, (rowsWin + 127) / 128), 256, 0, stream>>>(
        oc16, outT, nullptr, xo, zb, rowsWin, DIMC + V4C, DIMC, 0, 0, 0);
    resadd_kernel<<<(int)(((long)rowsX * DIMC + 255) / 256), 256, 0, stream>>>(x, xo);

    // ---- GLU feed-forward ----
    ln_kernel<<<rowsX, 256, 0, stream>>>(x, ln2_g + l * DIMC, ln2_b + l * DIMC, h16);
    gemm_bf16_wmma<<<dim3(2 * FFC / 64, (rowsX + 127) / 128), 256, 0, stream>>>(
        h16, gluT, glu_b + (size_t)l * 2 * FFC, ub, zb, rowsX, DIMC, 2 * FFC, 0, 0, 1);
    glu_kernel<<<(int)(((long)rowsX * FFC + 255) / 256), 256, 0, stream>>>(ub, tb16);
    gemm_bf16_wmma<<<dim3(DIMC / 64, (rowsX + 127) / 128), 256, 0, stream>>>(
        tb16, ffoT, ffo_b + (size_t)l * DIMC, x, zb, rowsX, FFC, DIMC,
        0, 0, 3 /* bias + accumulate into x */);
  }
}
